// Net_62113817035403
// MI455X (gfx1250) — compile-verified, matrix operand order checked by reading, not run on hardware
//
#include <hip/hip_runtime.h>
#include <hip/hip_bf16.h>

// ---------------------------------------------------------------------------
// Types for CDNA5 WMMA (wave32): D(16x16 f32) = A(16x32 bf16) x B(32x16 bf16) + C
// ---------------------------------------------------------------------------
typedef __attribute__((ext_vector_type(16))) __bf16        v16bf;
typedef __attribute__((ext_vector_type(8)))  float         v8f;
typedef __attribute__((ext_vector_type(4)))  unsigned int  v4u;
typedef __attribute__((ext_vector_type(8)))  int           v8i;
typedef __attribute__((ext_vector_type(4)))  int           v4i;

union FragU { v16bf v; v4u q[2]; };

#if defined(__has_builtin)
#if __has_builtin(__builtin_amdgcn_tensor_load_to_lds)
#define HAVE_TDM 1
#endif
#endif

// A-fragment (16x32, row-major, ld = K): lane row r = lane&15, kbase=(lane>>4)*8;
// elems 0..7 -> [r][kb..kb+7], elems 8..15 -> [r][kb+16..kb+23]. Two 16B loads.
// B-fragment (32x16 = K x N) from row-major [N,K]: same pattern with r -> col n.
__device__ __forceinline__ v16bf ld_frag_g(const __bf16* p0) {
  FragU f;
  f.q[0] = *(const v4u*)(p0);
  f.q[1] = *(const v4u*)(p0 + 16);
  return f.v;
}

__device__ __forceinline__ v8f wmma_bf16(v16bf a, v16bf b, v8f c) {
  return __builtin_amdgcn_wmma_f32_16x16x32_bf16(
      false, a, false, b, (short)0, c, false, false);
}

#ifdef HAVE_TDM
// Tensor Data Mover: DMA a 2D tile (rows x n_elems f32, row stride in elems)
// from global memory into LDS at lds_off. D# per CDNA5 ISA:
//  group0: [1:0]=count=1 | [63:32]=lds_addr | [120:64]=global_addr | [127:126]=type=2
//  group1: [17:16]=data_size=2 (4B) | tensor_dim0/1 | tile_dim0/1 | dim0_stride
// Groups 2/3 zero (2D tensor). Tracked by TENSORcnt.
// clang-23 toolchain form: 6 args (g0, g1, g2, g3, g4, cpol).
__device__ __forceinline__ void tdm_load_2d(unsigned lds_off, const float* gptr,
                                            unsigned n_elems, unsigned rows,
                                            unsigned row_stride_elems) {
  unsigned long long ga = (unsigned long long)(uintptr_t)gptr;
  v4u g0;
  g0[0] = 1u;                                         // count=1, user D#
  g0[1] = lds_off;                                    // lds_addr (bytes)
  g0[2] = (unsigned)(ga & 0xFFFFFFFFu);               // global_addr[31:0]
  g0[3] = (unsigned)((ga >> 32) & 0x01FFFFFFu) | (2u << 30);  // addr[56:32]|type=2
  v8i g1;
  g1[0] = (int)(2u << 16);                            // wg_mask=0, data_size=4B
  g1[1] = (int)((n_elems & 0xFFFFu) << 16);           // tensor_dim0[15:0]
  g1[2] = (int)(((n_elems >> 16) & 0xFFFFu) | ((rows & 0xFFFFu) << 16));
  g1[3] = (int)(((rows >> 16) & 0xFFFFu) | ((n_elems & 0xFFFFu) << 16)); // tile_dim0
  g1[4] = (int)(rows & 0xFFFFu);                      // tile_dim1, tile_dim2=0
  g1[5] = (int)row_stride_elems;                      // tensor_dim0_stride[31:0]
  g1[6] = 0;
  g1[7] = 0;
  v4i zg4 = {0, 0, 0, 0};
  v8i zg8 = {0, 0, 0, 0, 0, 0, 0, 0};
  __builtin_amdgcn_tensor_load_to_lds(g0, g1, zg4, zg4, zg8, 0);
}
#endif

// ---------------------------------------------------------------------------
// fp32 -> bf16 conversion
// ---------------------------------------------------------------------------
__global__ void cvt_bf16(const float* __restrict__ in, __bf16* __restrict__ out, int n) {
  int i = blockIdx.x * blockDim.x + threadIdx.x;
  if (i < n) out[i] = (__bf16)in[i];
}

// ---------------------------------------------------------------------------
// C[M,N] (f32) = A[M,K](bf16) @ W[N,K]^T (bf16) + bias[N].
// One wave computes 32x64 (2 M-tiles x 4 N-tiles): 6 fragment loads feed
// 8 WMMAs per K-step (A frags reused x4, B frags x2).
// M % 32 == 0, N % 64 == 0, K % 32 == 0 for all layers here.
// ---------------------------------------------------------------------------
__global__ void gemm_bf16(const __bf16* __restrict__ A, const __bf16* __restrict__ W,
                          const float* __restrict__ bias, float* __restrict__ C,
                          int M, int N, int K) {
  const int wave = (int)((blockIdx.x * blockDim.x + threadIdx.x) >> 5);
  const int lane = threadIdx.x & 31;
  const int ngroups = N >> 6;
  const int mtiles  = M >> 5;
  if (wave >= mtiles * ngroups) return;          // wave-uniform
  const int mt = wave / ngroups, ng = wave % ngroups;
  const int row0 = mt << 5, col0 = ng << 6;
  const int cl = lane & 15, kh = lane >> 4;

  v8f acc[2][4];
#pragma unroll
  for (int m = 0; m < 2; ++m)
#pragma unroll
    for (int j = 0; j < 4; ++j)
#pragma unroll
      for (int i = 0; i < 8; ++i) acc[m][j][i] = 0.0f;

  for (int k0 = 0; k0 < K; k0 += 32) {
    const int kb = k0 + (kh << 3);
    const v16bf a0 = ld_frag_g(A + (size_t)(row0 + cl) * K + kb);
    const v16bf a1 = ld_frag_g(A + (size_t)(row0 + 16 + cl) * K + kb);
#pragma unroll
    for (int j = 0; j < 4; ++j) {
      const v16bf b = ld_frag_g(W + (size_t)(col0 + (j << 4) + cl) * K + kb);
      acc[0][j] = wmma_bf16(a0, b, acc[0][j]);
      acc[1][j] = wmma_bf16(a1, b, acc[1][j]);
    }
  }
#pragma unroll
  for (int m = 0; m < 2; ++m)
#pragma unroll
    for (int j = 0; j < 4; ++j) {
      const int col = col0 + (j << 4) + cl;
      const float bv = bias[col];
#pragma unroll
      for (int i = 0; i < 8; ++i) {
        const int row = row0 + (m << 4) + i + (kh << 3);
        C[(size_t)row * N + col] = acc[m][j][i] + bv;
      }
    }
}

// ---------------------------------------------------------------------------
// Non-recurrent AdLIF scan (enc1): elementwise over (b,n), sequential in t.
// ---------------------------------------------------------------------------
__global__ void adlif_scan_ff(const float* __restrict__ cur,
                              const float* __restrict__ alpha,
                              const float* __restrict__ beta,
                              const float* __restrict__ a_p,
                              const float* __restrict__ bq,
                              float thr, __bf16* __restrict__ z_out,
                              int B, int T, int N) {
  const int idx = blockIdx.x * blockDim.x + threadIdx.x;
  if (idx >= B * N) return;
  const int b = idx / N, n = idx % N;
  const float al = alpha[n], be = beta[n], av = a_p[n], bv = bq[n];
  float u = 0.0f, w = 0.0f;
  const float* cp = cur + (size_t)b * T * N + n;
  __bf16* zp = z_out + (size_t)b * T * N + n;
  for (int t = 0; t < T; ++t) {
    const float it = cp[(size_t)t * N];
    float uu = al * u + (1.0f - al) * (it - w);
    const float zz = (uu > thr) ? 1.0f : 0.0f;
    uu *= (1.0f - zz);
    w = be * w + (1.0f - be) * (av * uu + bv * zz);
    u = uu;
    zp[(size_t)t * N] = (__bf16)zz;
  }
}

// ---------------------------------------------------------------------------
// Recurrent AdLIF scan. grid.x = 2 (batch groups of 16), block = 512 (16 waves).
// Per step: R(16xN) = Z_prev(16xN) @ V^T via WMMA (A-frags: ds_load_b128 from
// LDS spikes; B-frags: global_load_b128 from L2-resident bf16 V). The 16xN f32
// feedforward-current tile (row stride T*N) is DMA'd into LDS by the Tensor
// Data Mover (wave 0 issues, TENSORcnt-tracked), overlapped with the WMMAs.
// State (u,w) lives in accumulator-fragment layout in registers.
// ---------------------------------------------------------------------------
template <int NT>
__global__ void adlif_scan_rec(const float* __restrict__ cur,
                               const __bf16* __restrict__ V,
                               const float* __restrict__ alpha,
                               const float* __restrict__ beta,
                               const float* __restrict__ a_p,
                               const float* __restrict__ bq,
                               float thr, __bf16* __restrict__ z_out, int T) {
  constexpr int N = NT * 256;
  __shared__ __bf16 Zs[16 * N];     // previous-step spikes
  __shared__ float  CurS[16 * N];   // staged feedforward current for step t

  const int g    = blockIdx.x;      // batches [g*16, g*16+16)
  const int tid  = threadIdx.x;
  const int wv   = tid >> 5;        // 16 waves
  const int lane = tid & 31;
  const int cl   = lane & 15;
  const int kh   = lane >> 4;

  for (int i = tid; i < 16 * N; i += 512) Zs[i] = (__bf16)0.0f;

  int   col[NT];
  float al[NT], be[NT], av[NT], bv[NT];
  v8f   u[NT], w[NT];
#pragma unroll
  for (int j = 0; j < NT; ++j) {
    col[j] = (wv * NT + j) * 16 + cl;
    al[j] = alpha[col[j]]; be[j] = beta[col[j]];
    av[j] = a_p[col[j]];   bv[j] = bq[col[j]];
#pragma unroll
    for (int i = 0; i < 8; ++i) { u[j][i] = 0.0f; w[j][i] = 0.0f; }
  }
#ifdef HAVE_TDM
  const unsigned curs_lds = (unsigned)(uintptr_t)(void*)&CurS[0]; // LDS offset
#endif
  __syncthreads();

  const size_t rowbase = (size_t)(g * 16) * T;
  for (int t = 0; t < T; ++t) {
    // --- kick the DMA for this step's current tile, then overlap with WMMA ---
#ifdef HAVE_TDM
    if (wv == 0)
      tdm_load_2d(curs_lds, cur + (rowbase + (size_t)t) * N,
                  (unsigned)N, 16u, (unsigned)(T * N));
#else
    {   // fallback: coalesced copy, 32 threads per row
      const int r = tid >> 5, c0 = (tid & 31) * (N / 32);
      const float* gp = cur + (rowbase + (size_t)r * T + t) * N + c0;
#pragma unroll
      for (int c = 0; c < N / 32; c += 4)
        *(float4*)&CurS[r * N + c0 + c] = *(const float4*)(gp + c);
    }
#endif

    v8f acc[NT];
#pragma unroll
    for (int j = 0; j < NT; ++j)
#pragma unroll
      for (int i = 0; i < 8; ++i) acc[j][i] = 0.0f;

    for (int k0 = 0; k0 < N; k0 += 32) {
      const int kb = k0 + (kh << 3);
      FragU fa;
      fa.q[0] = *(const v4u*)&Zs[cl * N + kb];       // ds_load_b128
      fa.q[1] = *(const v4u*)&Zs[cl * N + kb + 16];
      const v16bf afrag = fa.v;
#pragma unroll
      for (int j = 0; j < NT; ++j) {
        const v16bf bfrag = ld_frag_g(V + (size_t)col[j] * N + kb);
        acc[j] = wmma_bf16(afrag, bfrag, acc[j]);
      }
    }
#ifdef HAVE_TDM
    if (wv == 0) __builtin_amdgcn_s_wait_tensorcnt(0);  // DMA landed in LDS
#endif
    __syncthreads();                  // CurS ready for all; done reading Zs

#pragma unroll
    for (int j = 0; j < NT; ++j) {
#pragma unroll
      for (int i = 0; i < 8; ++i) {
        const int brow = i + (kh << 3);               // batch within group
        const float it = CurS[brow * N + col[j]] + acc[j][i];
        float uu = al[j] * u[j][i] + (1.0f - al[j]) * (it - w[j][i]);
        const float zz = (uu > thr) ? 1.0f : 0.0f;
        uu *= (1.0f - zz);
        w[j][i] = be[j] * w[j][i] + (1.0f - be[j]) * (av[j] * uu + bv[j] * zz);
        u[j][i] = uu;
        const __bf16 zb = (__bf16)zz;
        z_out[(rowbase + (size_t)brow * T + t) * N + col[j]] = zb;
        Zs[brow * N + col[j]] = zb;
      }
    }
    __syncthreads();                  // Zs/CurS safe to overwrite next step
  }
}

// ---------------------------------------------------------------------------
// Readout: dots[row] = z4[row,:] . Wout + b ; one wave per row, shuffle reduce.
// ---------------------------------------------------------------------------
__global__ void readout_dot(const __bf16* __restrict__ z, const float* __restrict__ Wo,
                            const float* __restrict__ bo, float* __restrict__ dots,
                            int rows, int K) {
  const int wave = (int)((blockIdx.x * blockDim.x + threadIdx.x) >> 5);
  const int lane = threadIdx.x & 31;
  if (wave >= rows) return;
  const __bf16* zp = z + (size_t)wave * K;
  float s = 0.0f;
  for (int k = lane; k < K; k += 32) s += (float)zp[k] * Wo[k];
#pragma unroll
  for (int off = 16; off; off >>= 1) s += __shfl_down(s, off, 32);
  if (lane == 0) dots[wave] = s + bo[0];
}

__global__ void li_scan(const float* __restrict__ dots, const float* __restrict__ alpha,
                        float* __restrict__ out, int B, int T) {
  const int b = threadIdx.x;
  if (b >= B) return;
  const float al = alpha[0];
  float u = 0.0f;
  for (int t = 0; t < T; ++t) {
    u = al * u + (1.0f - al) * dots[(size_t)b * T + t];
    out[(size_t)b * T + t] = u;
  }
}

// ---------------------------------------------------------------------------
// Launch. d_in order (JAX pytree: dict keys sorted alphabetically):
//  0 inputs | dec1: 1 V 2 W 3 a 4 alpha 5 b 6 beta 7 bq
//  dec2: 8 V 9 W 10 a 11 alpha 12 b 13 beta 14 bq
//  enc1: 15 W 16 a 17 alpha 18 b 19 beta 20 bq
//  enc2: 21 V 22 W 23 a 24 alpha 25 b 26 beta 27 bq
//  out: 28 W 29 alpha 30 b
// ---------------------------------------------------------------------------
extern "C" void kernel_launch(void* const* d_in, const int* in_sizes, int n_in,
                              void* d_out, int out_size, void* d_ws, size_t ws_size,
                              hipStream_t stream) {
  constexpr int B = 32, T = 1000, D = 256, NB = 1024, NS = 512;
  constexpr int MT = B * T;                      // 32000 rows
  char* ws = (char*)d_ws;
  size_t off = 0;
  auto alloc = [&](size_t bytes) { char* p = ws + off; off = (off + bytes + 255) & ~(size_t)255; return p; };

  __bf16* xbf  = (__bf16*)alloc((size_t)MT * D  * 2);
  __bf16* z1   = (__bf16*)alloc((size_t)MT * NB * 2);
  __bf16* z2   = (__bf16*)alloc((size_t)MT * NS * 2);
  __bf16* z3   = (__bf16*)alloc((size_t)MT * NS * 2);
  __bf16* z4   = (__bf16*)alloc((size_t)MT * NB * 2);
  float*  cur  = (float*) alloc((size_t)MT * NB * 4);   // reused per layer
  __bf16* W1b  = (__bf16*)alloc((size_t)NB * D  * 2);
  __bf16* W2b  = (__bf16*)alloc((size_t)NS * NB * 2);
  __bf16* V2b  = (__bf16*)alloc((size_t)NS * NS * 2);
  __bf16* W3b  = (__bf16*)alloc((size_t)NS * NS * 2);
  __bf16* V3b  = (__bf16*)alloc((size_t)NS * NS * 2);
  __bf16* W4b  = (__bf16*)alloc((size_t)NB * NS * 2);
  __bf16* V4b  = (__bf16*)alloc((size_t)NB * NB * 2);

  const float* x = (const float*)d_in[0];
  auto F = [&](int i) { return (const float*)d_in[i]; };

  auto cvt = [&](const float* src, __bf16* dst, int n) {
    cvt_bf16<<<(n + 255) / 256, 256, 0, stream>>>(src, dst, n);
  };
  cvt(x,     xbf, MT * D);
  cvt(F(15), W1b, NB * D);
  cvt(F(22), W2b, NS * NB);
  cvt(F(21), V2b, NS * NS);
  cvt(F(2),  W3b, NS * NS);
  cvt(F(1),  V3b, NS * NS);
  cvt(F(9),  W4b, NB * NS);
  cvt(F(8),  V4b, NB * NB);

  auto gemm = [&](const __bf16* A, const __bf16* W, const float* bias, float* C,
                  int M, int N, int K) {
    const int waves = (M >> 5) * (N >> 6);
    gemm_bf16<<<(waves + 7) / 8, 256, 0, stream>>>(A, W, bias, C, M, N, K);
  };

  // enc1: no recurrence, thr = 0.1
  gemm(xbf, W1b, F(18), cur, MT, NB, D);
  adlif_scan_ff<<<(B * NB + 255) / 256, 256, 0, stream>>>(
      cur, F(17), F(19), F(16), F(20), 0.1f, z1, B, T, NB);

  // enc2: recurrent, N = 512
  gemm(z1, W2b, F(25), cur, MT, NS, NB);
  adlif_scan_rec<2><<<2, 512, 0, stream>>>(cur, V2b, F(24), F(26), F(23), F(27), 1.0f, z2, T);

  // dec1: recurrent, N = 512
  gemm(z2, W3b, F(5), cur, MT, NS, NS);
  adlif_scan_rec<2><<<2, 512, 0, stream>>>(cur, V3b, F(4), F(6), F(3), F(7), 1.0f, z3, T);

  // dec2: recurrent, N = 1024
  gemm(z3, W4b, F(12), cur, MT, NB, NS);
  adlif_scan_rec<4><<<2, 512, 0, stream>>>(cur, V4b, F(11), F(13), F(10), F(14), 1.0f, z4, T);

  // LI readout
  readout_dot<<<(MT + 7) / 8, 256, 0, stream>>>(z4, F(28), F(30), cur, MT, NB);
  li_scan<<<1, 32, 0, stream>>>(cur, F(29), (float*)d_out, B, T);
}
